// GptOssDecoderLayer_86595130622525
// MI455X (gfx1250) — matrix operand order repaired and checked by reading, not compiled, not run on hardware
//
#include <hip/hip_runtime.h>
#include <hip/hip_bf16.h>
#include <math.h>

// ---------------- problem constants ----------------
#define S_    2048
#define D_    2880
#define NH_   64
#define NKV_  8
#define HD_   64
#define I_    2880
#define E_    32
#define QKVD_ 5120          // (NH + 2*NKV) * HD
#define GU_   5760          // 2*I
#define REP_  (NH_ / NKV_)  // 8
#define LOG_THETA 11.918390573078392f   // ln(150000)

typedef __attribute__((ext_vector_type(16))) __bf16 v16bf;
typedef __attribute__((ext_vector_type(8)))  float  v8f;
typedef unsigned short ushort_t;

__device__ __forceinline__ unsigned short f2bf(float f) {
  union { float f; unsigned int u; } c; c.f = f;
  return (unsigned short)((c.u + 0x7FFFu + ((c.u >> 16) & 1u)) >> 16);
}

union FragBF {
  v16bf v;
  uint4 q[2];
  unsigned short u16[16];
  unsigned int   u32[8];
};

__device__ __forceinline__ v8f v8f_zero() {
  v8f z = {0.f, 0.f, 0.f, 0.f, 0.f, 0.f, 0.f, 0.f};
  return z;
}

__device__ __forceinline__ v8f wmma_bf16(v16bf a, v16bf b, v8f c) {
  return __builtin_amdgcn_wmma_f32_16x16x32_bf16(false, a, false, b,
                                                 (short)0, c, false, false);
}

// CDNA5 async global->LDS copy, 16 bytes per lane (ASYNCcnt tracked).
__device__ __forceinline__ void async_ld16_off(unsigned loff, const void* gptr) {
  asm volatile("global_load_async_to_lds_b128 %0, %1, off"
               :: "v"(loff), "v"(gptr) : "memory");
}
__device__ __forceinline__ void wait_async0() {
  asm volatile("s_wait_asynccnt 0x0" ::: "memory");
}
__device__ __forceinline__ unsigned lds_off(const void* p) {
  // shared-aperture generic pointers carry the LDS byte offset in [31:0]
  return (unsigned)(size_t)p;
}
__device__ __forceinline__ int swap12(int c) {  // fragment chunk permutation
  return (c == 1) ? 2 : (c == 2) ? 1 : c;
}

// ============================================================
// WMMA GEMM:  C[M,N] = A[M,K](bf16) * B[N,K](bf16)^T + bias[N]
// 256 threads (8 waves, 4x2), tile 128x128, K-step 32,
// double-buffered async-to-LDS staging with hoisted addresses.
// ============================================================
__global__ __launch_bounds__(256)
void gemm_bf16_kernel(const ushort_t* __restrict__ A,
                      const ushort_t* __restrict__ B,
                      const float* __restrict__ bias, float* __restrict__ C,
                      int M, int N, int K) {
  __shared__ ushort_t As[2][128][32];   // natural rows
  __shared__ ushort_t Bs[2][128][32];   // natural rows
  const int tid  = threadIdx.x;
  const int wave = tid >> 5, lane = tid & 31;
  const int hi   = lane >> 4, lr = lane & 15;
  const int wr   = wave >> 1, wc = wave & 1;
  const int m0 = blockIdx.y * 128, n0 = blockIdx.x * 128;
  const unsigned bufStride = 128 * 32 * 2;  // 8 KB per stage

  // zero OOB B rows once (edge tiles only); stager never writes them.
  if (n0 + 128 > N) {
    for (int t = tid; t < 128 * 4; t += 256) {
      int r = t >> 2, c = t & 3;
      if (n0 + r >= N) {
        uint4 z = {0u, 0u, 0u, 0u};
        *(uint4*)&Bs[0][r][c * 8] = z;
        *(uint4*)&Bs[1][r][c * 8] = z;
      }
    }
  }

  // hoisted per-thread staging addresses: 2 A chunks + 2 B chunks,
  // global side advances 64 B per K-step.
  const int r0 = tid >> 2, c0 = tid & 3;            // t = tid
  const int r1 = (tid + 256) >> 2, c1 = tid & 3;    // t = tid + 256
  const ushort_t* aG0 = &A[(size_t)(m0 + r0) * K + c0 * 8];
  const ushort_t* aG1 = &A[(size_t)(m0 + r1) * K + c1 * 8];
  const ushort_t* bG0 = &B[(size_t)(n0 + r0) * K + c0 * 8];
  const ushort_t* bG1 = &B[(size_t)(n0 + r1) * K + c1 * 8];
  const unsigned aL0 = lds_off(&As[0][r0][c0 * 8]);
  const unsigned aL1 = lds_off(&As[0][r1][c1 * 8]);
  const unsigned bL0 = lds_off(&Bs[0][r0][c0 * 8]);
  const unsigned bL1 = lds_off(&Bs[0][r1][c1 * 8]);
  const bool bv0 = (n0 + r0) < N;
  const bool bv1 = (n0 + r1) < N;

  auto stage = [&](int buf) {
    unsigned o = buf ? bufStride : 0u;
    async_ld16_off(aL0 + o, aG0);
    async_ld16_off(aL1 + o, aG1);
    if (bv0) async_ld16_off(bL0 + o, bG0);
    if (bv1) async_ld16_off(bL1 + o, bG1);
    aG0 += 32; aG1 += 32; bG0 += 32; bG1 += 32;
  };

  stage(0);
  v8f acc[2][4];
#pragma unroll
  for (int im = 0; im < 2; ++im)
#pragma unroll
    for (int j = 0; j < 4; ++j) acc[im][j] = v8f_zero();

  const int nsteps = K / 32;
  wait_async0();
  __syncthreads();

  for (int s = 0; s < nsteps; ++s) {
    const int buf = s & 1;
    if (s + 1 < nsteps) stage(buf ^ 1);

    FragBF a[2], b[4];
#pragma unroll
    for (int im = 0; im < 2; ++im) {
      const ushort_t* p = &As[buf][wr * 32 + im * 16 + lr][hi * 16];
      a[im].q[0] = *(const uint4*)p;
      a[im].q[1] = *(const uint4*)(p + 8);
    }
#pragma unroll
    for (int j = 0; j < 4; ++j) {
      const ushort_t* p = &Bs[buf][wc * 64 + j * 16 + lr][hi * 16];
      b[j].q[0] = *(const uint4*)p;
      b[j].q[1] = *(const uint4*)(p + 8);
    }
#pragma unroll
    for (int im = 0; im < 2; ++im)
#pragma unroll
      for (int j = 0; j < 4; ++j)
        acc[im][j] = wmma_bf16(a[im].v, b[j].v, acc[im][j]);

    wait_async0();
    __syncthreads();
  }

#pragma unroll
  for (int im = 0; im < 2; ++im)
#pragma unroll
    for (int j = 0; j < 4; ++j) {
      int n = n0 + wc * 64 + j * 16 + lr;
      if (n >= N) continue;
      float bb = bias ? bias[n] : 0.f;
#pragma unroll
      for (int r = 0; r < 8; ++r) {
        int m = m0 + wr * 32 + im * 16 + r + 8 * hi;
        C[(size_t)m * N + n] = acc[im][j][r] + bb;
      }
    }
}

// ============================================================
// f32 -> bf16 bulk convert
// ============================================================
__global__ __launch_bounds__(256)
void cvt_bf16_kernel(const float* __restrict__ x, ushort_t* __restrict__ y, int n) {
  int i = blockIdx.x * 256 + threadIdx.x;
  if (i < n) y[i] = f2bf(x[i]);
}

// ============================================================
// RMSNorm -> bf16
// ============================================================
__global__ __launch_bounds__(256)
void rmsnorm_kernel(const float* __restrict__ x, const float* __restrict__ w,
                    ushort_t* __restrict__ y) {
  __shared__ float red[256];
  const int s = blockIdx.x;
  float acc = 0.f;
  for (int d = threadIdx.x; d < D_; d += 256) {
    float v = x[(size_t)s * D_ + d];
    acc += v * v;
  }
  red[threadIdx.x] = acc;
  __syncthreads();
  for (int off = 128; off > 0; off >>= 1) {
    if (threadIdx.x < off) red[threadIdx.x] += red[threadIdx.x + off];
    __syncthreads();
  }
  float r = rsqrtf(red[0] / (float)D_ + 1e-5f);
  for (int d = threadIdx.x; d < D_; d += 256)
    y[(size_t)s * D_ + d] = f2bf(x[(size_t)s * D_ + d] * r * w[d]);
}

// ============================================================
// RoPE + pack:  q -> [NH][S][HD], k -> [NKV][S][HD],
//               v -> [NKV][HD][S] (pre-transposed for PV fragments)
// ============================================================
__global__ __launch_bounds__(256)
void rope_pack_kernel(const int* __restrict__ pos, const float* __restrict__ qkv,
                      ushort_t* __restrict__ qb, ushort_t* __restrict__ kb,
                      ushort_t* __restrict__ vbt) {
  const int s = blockIdx.x;
  const float p = (float)pos[s];
  const size_t rowb = (size_t)s * QKVD_;
  for (int t = threadIdx.x; t < NH_ * HD_; t += 256) {
    int h = t >> 6, d = t & 63, i = d & 31;
    float invf = __expf(-((float)(2 * i) / (float)HD_) * LOG_THETA);
    float sn, cs;
    __sincosf(p * invf, &sn, &cs);
    float x1 = qkv[rowb + h * HD_ + i];
    float x2 = qkv[rowb + h * HD_ + 32 + i];
    float val = (d < 32) ? (x1 * cs - x2 * sn) : (x2 * cs + x1 * sn);
    qb[((size_t)h * S_ + s) * HD_ + d] = f2bf(val);
  }
  for (int t = threadIdx.x; t < NKV_ * HD_; t += 256) {
    int h = t >> 6, d = t & 63, i = d & 31;
    float invf = __expf(-((float)(2 * i) / (float)HD_) * LOG_THETA);
    float sn, cs;
    __sincosf(p * invf, &sn, &cs);
    float x1 = qkv[rowb + NH_ * HD_ + h * HD_ + i];
    float x2 = qkv[rowb + NH_ * HD_ + h * HD_ + 32 + i];
    float val = (d < 32) ? (x1 * cs - x2 * sn) : (x2 * cs + x1 * sn);
    kb[((size_t)h * S_ + s) * HD_ + d] = f2bf(val);
    float vv = qkv[rowb + NH_ * HD_ + NKV_ * HD_ + h * HD_ + d];
    vbt[((size_t)h * HD_ + d) * S_ + s] = f2bf(vv);  // transposed
  }
}

// ============================================================
// Flash-style causal GQA attention; WMMA for QK^T and PV.
// One block = (head, 64 query rows), 128 threads (4 waves).
// ============================================================
__global__ __launch_bounds__(128)
void attn_kernel(const ushort_t* __restrict__ qbuf,
                 const ushort_t* __restrict__ kbuf,
                 const ushort_t* __restrict__ vbt,
                 ushort_t* __restrict__ obf) {
  __shared__ ushort_t Qs[64][64];   // natural rows, [q][d]
  __shared__ ushort_t Ks[64][64];   // [key][d]
  __shared__ ushort_t Vt[64][64];   // [d][key]
  __shared__ float Sc[64][68];      // scores f32; P bf16 overlays bytes [0,128)
  __shared__ float Os[64][65];
  __shared__ float rowm[64], rowl[64];

  const int h   = blockIdx.x;
  const int qb0 = blockIdx.y * 64;
  const int kvh = h / REP_;
  const int tid  = threadIdx.x;
  const int wave = tid >> 5, lane = tid & 31;
  const int hi   = lane >> 4, lr = lane & 15;
  const int am   = wave * 16 + lr;
  const float scale = 0.125f;  // 64^-0.5

  // stage Q (natural layout) via async copies
  for (int t = tid; t < 512; t += 128) {
    int r = t >> 3, c = t & 7;
    int g = c >> 2, dc = c;
    async_ld16_off(lds_off(&Qs[r][dc * 8]),
                   &qbuf[((size_t)h * S_ + qb0 + r) * HD_ + c * 8]);
  }
  for (int t = tid; t < 64 * 64; t += 128) Os[t >> 6][t & 63] = 0.f;
  if (tid < 64) { rowm[tid] = -INFINITY; rowl[tid] = 0.f; }

  // hoisted K/V staging addresses (4 chunks each per thread)
  const ushort_t* kG[4];
  const ushort_t* vG[4];
  unsigned kL[4], vL[4];
#pragma unroll
  for (int j = 0; j < 4; ++j) {
    int t = tid + j * 128;
    int r = t >> 3, c = t & 7;
    kG[j] = &kbuf[((size_t)kvh * S_ + r) * HD_ + c * 8];
    vG[j] = &vbt[((size_t)kvh * HD_ + r) * S_ + c * 8];
    kL[j] = lds_off(&Ks[r][c * 8]);
    vL[j] = lds_off(&Vt[r][c * 8]);
  }

  wait_async0();
  __syncthreads();

  FragBF aq[2];
#pragma unroll
  for (int ks = 0; ks < 2; ++ks) {
    const ushort_t* p = &Qs[am][ks * 32 + hi * 16];
    aq[ks].q[0] = *(const uint4*)p;
    aq[ks].q[1] = *(const uint4*)(p + 8);
  }

  for (int kb0 = 0; kb0 <= qb0; kb0 += 64) {
#pragma unroll
    for (int j = 0; j < 4; ++j) {
      async_ld16_off(kL[j], kG[j]);
      async_ld16_off(vL[j], vG[j]);
      kG[j] += 64 * HD_;   // next 64 keys (row-major K)
      vG[j] += 64;         // next 64 keys (S-major V^T)
    }
    wait_async0();
    __syncthreads();

    // scores = Q K^T * scale
#pragma unroll
    for (int nb = 0; nb < 4; ++nb) {
      v8f acc = v8f_zero();
#pragma unroll
      for (int ks = 0; ks < 2; ++ks) {
        FragBF bfr;
        const ushort_t* p = &Ks[nb * 16 + lr][ks * 32 + hi * 16];
        bfr.q[0] = *(const uint4*)p;
        bfr.q[1] = *(const uint4*)(p + 8);
        acc = wmma_bf16(aq[ks].v, bfr.v, acc);
      }
#pragma unroll
      for (int r = 0; r < 8; ++r)
        Sc[wave * 16 + r + 8 * hi][nb * 16 + lr] = acc[r] * scale;
    }
    __syncthreads();

    // online softmax; write P bf16 in natural key order over the score row
    if (tid < 64) {
      int qpos = qb0 + tid;
      int jmax = qpos - kb0 + 1;
      if (jmax > 64) jmax = 64;
      float* srow = &Sc[tid][0];
      ushort_t* prow = (ushort_t*)srow;
      float mx = -INFINITY;
      for (int j = 0; j < jmax; ++j) mx = fmaxf(mx, srow[j]);
      float mold = rowm[tid];
      float mnew = fmaxf(mold, mx);
      float alpha = __expf(mold - mnew);
      float lsum = 0.f;
#pragma unroll
      for (int g = 0; g < 2; ++g) {
        float tmp[32];
#pragma unroll
        for (int j = 0; j < 32; ++j) {
          int jj = g * 32 + j;
          float pv = (jj < jmax) ? __expf(srow[jj] - mnew) : 0.f;
          tmp[j] = pv;
          lsum += pv;
        }
#pragma unroll
        for (int c = 0; c < 4; ++c) {
          int dc = c;
#pragma unroll
          for (int j = 0; j < 8; j += 2) {
            unsigned lo = f2bf(tmp[c * 8 + j]);
            unsigned h2 = f2bf(tmp[c * 8 + j + 1]);
            *(unsigned int*)&prow[g * 32 + dc * 8 + j] = lo | (h2 << 16);
          }
        }
      }
      rowl[tid] = rowl[tid] * alpha + lsum;
      rowm[tid] = mnew;
      for (int d = 0; d < HD_; ++d) Os[tid][d] *= alpha;
    }
    __syncthreads();

    // O += P V
#pragma unroll
    for (int nb = 0; nb < 4; ++nb) {
      v8f acc = v8f_zero();
#pragma unroll
      for (int ks = 0; ks < 2; ++ks) {
        FragBF ap, bv;
        const ushort_t* pp = (const ushort_t*)&Sc[am][0] + ks * 32 + hi * 16;
        ap.q[0] = *(const uint4*)pp;
        ap.q[1] = *(const uint4*)(pp + 8);
        const ushort_t* pv = &Vt[nb * 16 + lr][ks * 32 + hi * 16];
        bv.q[0] = *(const uint4*)pv;
        bv.q[1] = *(const uint4*)(pv + 8);
        acc = wmma_bf16(ap.v, bv.v, acc);
      }
#pragma unroll
      for (int r = 0; r < 8; ++r)
        Os[wave * 16 + r + 8 * hi][nb * 16 + lr] += acc[r];
    }
    __syncthreads();
  }

  if (tid < 64) {
    float inv = 1.f / rowl[tid];
    for (int d = 0; d < HD_; ++d)
      obf[((size_t)(qb0 + tid)) * (NH_ * HD_) + h * HD_ + d] =
          f2bf(Os[tid][d] * inv);
  }
}

// ============================================================
// small elementwise kernels
// ============================================================
__global__ void residual_add_kernel(const float* __restrict__ a,
                                    const float* __restrict__ b,
                                    float* __restrict__ y, int n) {
  int i = blockIdx.x * blockDim.x + threadIdx.x;
  if (i < n) y[i] = a[i] + b[i];
}

__global__ void router_topk_kernel(const float* __restrict__ logits,
                                   float* __restrict__ gate) {
  int s = blockIdx.x * blockDim.x + threadIdx.x;
  if (s >= S_) return;
  float v[E_];
  for (int e = 0; e < E_; ++e) v[e] = logits[(size_t)s * E_ + e];
  float top[4];
  for (int k = 0; k < 4; ++k) {
    float mx = -INFINITY; int mi = 0;
    for (int e = 0; e < E_; ++e)
      if (v[e] > mx) { mx = v[e]; mi = e; }
    top[k] = mx;
    v[mi] = -INFINITY;
  }
  float mx = top[0], den = 0.f;
  for (int k = 0; k < 4; ++k) den += __expf(top[k] - mx);
  float sum = 0.f;
  for (int k = 0; k < 4; ++k) sum += __expf(top[k] - mx) / den;
  gate[s] = sum;
}

__global__ void silu_gate_kernel(const float* __restrict__ gu,
                                 ushort_t* __restrict__ x, int n) {
  int idx = blockIdx.x * blockDim.x + threadIdx.x;
  if (idx >= n) return;
  int s = idx / I_, i = idx % I_;
  float g  = gu[(size_t)s * GU_ + i];
  float up = gu[(size_t)s * GU_ + I_ + i];
  x[idx] = f2bf(g * (up / (1.f + __expf(-up))));
}

__global__ void scale_rows_kernel(float* __restrict__ y,
                                  const float* __restrict__ gate, int n) {
  int i = blockIdx.x * blockDim.x + threadIdx.x;
  if (i < n) y[i] *= gate[i / D_];
}

// ============================================================
// orchestration
// ============================================================
static inline void launch_gemm(const ushort_t* A, const ushort_t* B,
                               const float* bias, float* C, int M, int N, int K,
                               hipStream_t st) {
  dim3 grid((N + 127) / 128, M / 128);
  gemm_bf16_kernel<<<grid, dim3(256), 0, st>>>(A, B, bias, C, M, N, K);
}
static inline void launch_cvt(const float* x, ushort_t* y, int n, hipStream_t st) {
  cvt_bf16_kernel<<<(n + 255) / 256, 256, 0, st>>>(x, y, n);
}

extern "C" void kernel_launch(void* const* d_in, const int* in_sizes, int n_in,
                              void* d_out, int out_size, void* d_ws, size_t ws_size,
                              hipStream_t stream) {
  const int*   positions = (const int*)  d_in[0];
  const float* hidden    = (const float*)d_in[1];
  const float* qkv_w     = (const float*)d_in[2];
  const float* qkv_b     = (const float*)d_in[3];
  const float* o_w       = (const float*)d_in[4];
  const float* o_b       = (const float*)d_in[5];
  const float* ln1_w     = (const float*)d_in[6];
  const float* ln2_w     = (const float*)d_in[7];
  const float* router_w  = (const float*)d_in[8];
  const float* router_b  = (const float*)d_in[9];
  const float* gu_w      = (const float*)d_in[10];
  const float* gu_b      = (const float*)d_in[11];
  const float* dn_w      = (const float*)d_in[12];
  const float* dn_b      = (const float*)d_in[13];

  float* out_main = (float*)d_out;               // (S, D)
  float* residual = out_main + (size_t)S_ * D_;  // (S, D)

  // ---- workspace layout ----
  float* region0   = (float*)d_ws;                       // S*GU : qkv f32 -> gu f32
  float* attn_proj = region0 + (size_t)S_ * GU_;         // S*D
  float* logits    = attn_proj + (size_t)S_ * D_;        // S*E
  float* gatesc    = logits + (size_t)S_ * E_;           // S
  ushort_t* hbf  = (ushort_t*)(gatesc + S_);             // S*D
  ushort_t* qbf  = hbf + (size_t)S_ * D_;                // NH*S*HD
  ushort_t* kbf  = qbf + (size_t)NH_ * S_ * HD_;         // NKV*S*HD
  ushort_t* vbt  = kbf + (size_t)NKV_ * S_ * HD_;        // NKV*HD*S
  ushort_t* obf  = vbt + (size_t)NKV_ * HD_ * S_;        // S*(NH*HD)
  ushort_t* xbf  = obf + (size_t)S_ * NH_ * HD_;         // S*I
  ushort_t* wqkv = xbf + (size_t)S_ * I_;                // QKVD*D
  ushort_t* wo   = wqkv + (size_t)QKVD_ * D_;            // D*(NH*HD)
  ushort_t* wgu  = wo + (size_t)D_ * NH_ * HD_;          // GU*D
  ushort_t* wdn  = wgu + (size_t)GU_ * D_;               // D*I
  ushort_t* wr   = wdn + (size_t)D_ * I_;                // E*D

  const int eltN = S_ * D_;

  // weight conversions (every call; no cross-call state)
  launch_cvt(qkv_w, wqkv, QKVD_ * D_, stream);
  launch_cvt(o_w, wo, D_ * NH_ * HD_, stream);
  launch_cvt(gu_w, wgu, GU_ * D_, stream);
  launch_cvt(dn_w, wdn, D_ * I_, stream);
  launch_cvt(router_w, wr, E_ * D_, stream);

  // 1) h = rmsnorm(hidden) -> bf16
  rmsnorm_kernel<<<S_, 256, 0, stream>>>(hidden, ln1_w, hbf);
  // 2) qkv = h @ qkv_w^T + b  (f32 out)
  launch_gemm(hbf, wqkv, qkv_b, region0, S_, QKVD_, D_, stream);
  // 3) RoPE + pack
  rope_pack_kernel<<<S_, 256, 0, stream>>>(positions, region0, qbf, kbf, vbt);
  // 4) attention -> obf (bf16)
  attn_kernel<<<dim3(NH_, S_ / 64), dim3(128), 0, stream>>>(qbf, kbf, vbt, obf);
  // 5) attn_out = o @ o_w^T + o_b
  launch_gemm(obf, wo, o_b, attn_proj, S_, D_, NH_ * HD_, stream);
  // 6) residual = attn_out + hidden
  residual_add_kernel<<<(eltN + 255) / 256, 256, 0, stream>>>(attn_proj, hidden,
                                                              residual, eltN);
  // 7) h2 = rmsnorm(residual) -> bf16
  rmsnorm_kernel<<<S_, 256, 0, stream>>>(residual, ln2_w, hbf);
  // 8) router logits
  launch_gemm(hbf, wr, router_b, logits, S_, E_, D_, stream);
  // 9) top-4 softmax sum
  router_topk_kernel<<<(S_ + 255) / 256, 256, 0, stream>>>(logits, gatesc);
  // 10) gate_up = h2 @ gu_w^T + gu_b
  launch_gemm(hbf, wgu, gu_b, region0, S_, GU_, D_, stream);
  // 11) x = gate * silu(up) -> bf16
  {
    int n = S_ * I_;
    silu_gate_kernel<<<(n + 255) / 256, 256, 0, stream>>>(region0, xbf, n);
  }
  // 12) expert_out = x @ dn_w^T + dn_b -> out
  launch_gemm(xbf, wdn, dn_b, out_main, S_, D_, I_, stream);
  // 13) out *= router gate
  scale_rows_kernel<<<(eltN + 255) / 256, 256, 0, stream>>>(out_main, gatesc, eltN);
}
